// DualGNNModel_44504451121832
// MI455X (gfx1250) — matrix operand order repaired and checked by reading, not compiled, not run on hardware
//
#include <hip/hip_runtime.h>

// ---------------------------------------------------------------------------
// DualGNN (2x 3-layer GCN encoders + MLP head) for MI455X / gfx1250 (wave32).
//
// Roofline: edge gather/scatter dominates (~5 GB irregular traffic over the
// whole model vs ~8 GFLOP of dense work). Dense X@W transforms are mapped to
// v_wmma_f32_16x16x32_f16 (f16 operands, f32 accumulate); scatter uses one
// wave per edge with float4 lanes + global_atomic_add_f32 (destinations live
// in the 192MB L2: N*128*4 = 25.6MB).
// ---------------------------------------------------------------------------

typedef __attribute__((ext_vector_type(16))) _Float16 v16h;
typedef __attribute__((ext_vector_type(8)))  float    v8f;

#define DHF 128   // hidden feature width (DH)

// ---- degree accumulate: deg[idx[e]] += 1 ----------------------------------
__global__ void k_deg(const int* __restrict__ idx, float* __restrict__ deg, int e) {
    int t = blockIdx.x * blockDim.x + threadIdx.x;
    if (t < e) atomicAdd(&deg[idx[t]], 1.0f);
}

// ---- deg -> clip(deg,1)^-0.5 in place -------------------------------------
__global__ void k_norm(float* __restrict__ deg, int n) {
    int t = blockIdx.x * blockDim.x + threadIdx.x;
    if (t < n) {
        float d = deg[t];
        d = d < 1.0f ? 1.0f : d;
        deg[t] = rsqrtf(d);
    }
}

// ---- Hout[n,128] = diag(nsrc) * Hin[n,K] @ W[K,128]  (WMMA f16->f32) ------
// block = 256 threads = 8 waves; each wave computes a 16x128 output slab.
template<int K>
__global__ void k_gemm_wmma(const float* __restrict__ Hin,
                            const float* __restrict__ W,
                            const float* __restrict__ nsrc,
                            float* __restrict__ Hout, int n) {
    // W staged transposed as f16: Wt[col*K + k] = W[k*128 + col]
    __shared__ _Float16 Wt[K * DHF];
    for (int i = threadIdx.x; i < K * DHF; i += blockDim.x) {
        int k = i >> 7;          // i / 128  (coalesced global read)
        int c = i & 127;
        Wt[c * K + k] = (_Float16)W[i];
    }
    __syncthreads();

    const int wave = threadIdx.x >> 5;
    const int lane = threadIdx.x & 31;
    const int hi   = lane >> 4;          // lane half (ISA K-split)
    const int lm   = lane & 15;

    const int row0 = (blockIdx.x * 8 + wave) * 16;
    if (row0 >= n) return;               // wave-uniform exit

    const int row   = row0 + lm;         // A-matrix row for this lane
    const bool valid = row < n;
    float scale = 0.0f;
    const float* arow = Hin;
    if (valid) { scale = nsrc[row]; arow = Hin + (size_t)row * K; }

    v8f acc[8];
    #pragma unroll
    for (int nt = 0; nt < 8; nt++) acc[nt] = (v8f){0,0,0,0,0,0,0,0};

    #pragma unroll
    for (int ks = 0; ks < K / 32; ks++) {
        const int kbase = ks * 32;
        // A fragment: 16x32 f16 layout -> two contiguous 8-wide K chunks
        v16h a;
        if (valid) {
            #pragma unroll
            for (int i = 0; i < 8; i++)
                a[i] = (_Float16)(arow[kbase + 16 * hi + i] * scale);
            #pragma unroll
            for (int i = 0; i < 8; i++)
                a[8 + i] = (_Float16)(arow[kbase + 16 * hi + 8 + i] * scale);
        } else {
            #pragma unroll
            for (int i = 0; i < 16; i++) a[i] = (_Float16)0.0f;
        }
        #pragma unroll
        for (int nt = 0; nt < 8; nt++) {
            const int col = nt * 16 + lm;
            // B fragment: b[i] = W[kbase + 16*hi + i, col] -> contiguous in Wt
            v16h b = *(const v16h*)&Wt[col * K + kbase + 16 * hi];
            acc[nt] = __builtin_amdgcn_wmma_f32_16x16x32_f16(
                false, a, false, b, (short)0, acc[nt], false, false);
        }
    }

    // D layout: acc[nt][r] = D[row0 + r + 8*hi, nt*16 + lm]
    #pragma unroll
    for (int nt = 0; nt < 8; nt++) {
        const int col = nt * 16 + lm;
        #pragma unroll
        for (int r = 0; r < 8; r++) {
            const int orow = row0 + r + 8 * hi;
            if (orow < n) Hout[(size_t)orow * DHF + col] = acc[nt][r];
        }
    }
}

// ---- edge scatter: M[dst[e],:] += H[src[e],:]  (one wave per edge) --------
__global__ void k_scatter(const float* __restrict__ H, const int* __restrict__ src,
                          const int* __restrict__ dst, float* __restrict__ M, int e) {
    int t = blockIdx.x * blockDim.x + threadIdx.x;
    int edge = t >> 5;
    if (edge >= e) return;
    int l = t & 31;
    int s = src[edge];
    int d = dst[edge];
    const float4 v = *(const float4*)(H + (size_t)s * DHF + l * 4);
    float* p = M + (size_t)d * DHF + l * 4;
    atomicAdd(p + 0, v.x);
    atomicAdd(p + 1, v.y);
    atomicAdd(p + 2, v.z);
    atomicAdd(p + 3, v.w);
}

// ---- h = relu(M * ndst[:,None] + b) in place ------------------------------
__global__ void k_bias_relu(float* __restrict__ M, const float* __restrict__ ndst,
                            const float* __restrict__ b, int n) {
    int t = blockIdx.x * blockDim.x + threadIdx.x;
    if (t < n * DHF) {
        int row = t >> 7;
        int c   = t & (DHF - 1);
        float v = M[t] * ndst[row] + b[c];
        M[t] = v > 0.0f ? v : 0.0f;
    }
}

// ---- per-graph sum pooling + counts (one wave per node) -------------------
__global__ void k_pool(const float* __restrict__ H, const int* __restrict__ gid,
                       float* __restrict__ pooled, float* __restrict__ cnt, int n) {
    int t = blockIdx.x * blockDim.x + threadIdx.x;
    int node = t >> 5;
    if (node >= n) return;
    int l = t & 31;
    int g = gid[node];
    const float4 v = *(const float4*)(H + (size_t)node * DHF + l * 4);
    float* p = pooled + (size_t)g * DHF + l * 4;
    atomicAdd(p + 0, v.x);
    atomicAdd(p + 1, v.y);
    atomicAdd(p + 2, v.z);
    atomicAdd(p + 3, v.w);
    if (l == 0) atomicAdd(&cnt[g], 1.0f);
}

// ---- MLP head: [mean_su | mean_sv | gf] -> 128 relu -> 64 relu -> 1 -------
__global__ void k_mlp(const float* __restrict__ psu, const float* __restrict__ csu,
                      const float* __restrict__ psv, const float* __restrict__ csv,
                      const float* __restrict__ gf, int ngf,
                      const float* __restrict__ W0, const float* __restrict__ b0,
                      const float* __restrict__ W1, const float* __restrict__ b1,
                      const float* __restrict__ W2, const float* __restrict__ b2,
                      float* __restrict__ out) {
    int g = blockIdx.x;
    int j = threadIdx.x;                       // 128 threads
    __shared__ float comb[2 * DHF + 16];
    __shared__ float h1[128];
    __shared__ float h2[64];

    float cu = csu[g]; cu = cu < 1.0f ? 1.0f : cu;
    float cv = csv[g]; cv = cv < 1.0f ? 1.0f : cv;
    comb[j]        = psu[(size_t)g * DHF + j] / cu;
    comb[DHF + j]  = psv[(size_t)g * DHF + j] / cv;
    if (j < ngf) comb[2 * DHF + j] = gf[g * ngf + j];
    __syncthreads();

    const int d0 = 2 * DHF + ngf;
    float a = b0[j];
    for (int k = 0; k < d0; k++) a += comb[k] * W0[k * 128 + j];
    h1[j] = a > 0.0f ? a : 0.0f;
    __syncthreads();

    if (j < 64) {
        float a2 = b1[j];
        for (int k = 0; k < 128; k++) a2 += h1[k] * W1[k * 64 + j];
        h2[j] = a2 > 0.0f ? a2 : 0.0f;
    }
    __syncthreads();

    if (j == 0) {
        float a3 = b2[0];
        for (int k = 0; k < 64; k++) a3 += h2[k] * W2[k];
        out[g] = a3;
    }
}

// ---------------------------------------------------------------------------
static inline size_t align256(size_t x) { return (x + 255) & ~(size_t)255; }

extern "C" void kernel_launch(void* const* d_in, const int* in_sizes, int n_in,
                              void* d_out, int out_size, void* d_ws, size_t ws_size,
                              hipStream_t stream) {
    (void)n_in; (void)out_size; (void)ws_size;
    const int N   = in_sizes[7];           // nodes (gid length)
    const int E   = in_sizes[3];           // edges
    const int G   = in_sizes[7] > 0 ? in_sizes[2] / 4 : 0;  // graphs (gf is [G,4])
    const int NGF = G > 0 ? in_sizes[2] / G : 4;

    // ---- workspace carve-out ----
    char* w = (char*)d_ws;
    size_t off = 0;
    auto take = [&](size_t bytes) { void* p = w + off; off += align256(bytes); return p; };
    float* M0      = (float*)take((size_t)N * DHF * 4);
    float* M1      = (float*)take((size_t)N * DHF * 4);
    float* nsrc    = (float*)take((size_t)N * 4);
    float* ndst    = (float*)take((size_t)N * 4);
    float* pool_su = (float*)take((size_t)G * DHF * 4);
    float* pool_sv = (float*)take((size_t)G * DHF * 4);
    float* cnt_su  = (float*)take((size_t)G * 4);
    float* cnt_sv  = (float*)take((size_t)G * 4);

    for (int enc = 0; enc < 2; enc++) {
        const float* x   = (const float*)d_in[enc == 0 ? 0 : 1];
        const int*   src = (const int*)  d_in[enc == 0 ? 3 : 5];
        const int*   dst = (const int*)  d_in[enc == 0 ? 4 : 6];
        const int*   gid = (const int*)  d_in[enc == 0 ? 7 : 8];
        const float* W0  = (const float*)d_in[enc == 0 ? 9  : 13];
        const float* W1  = (const float*)d_in[enc == 0 ? 10 : 14];
        const float* W2  = (const float*)d_in[enc == 0 ? 11 : 15];
        const float* bb  = (const float*)d_in[enc == 0 ? 12 : 16];
        float* pooled    = enc == 0 ? pool_su : pool_sv;
        float* cnt       = enc == 0 ? cnt_su  : cnt_sv;

        // degrees -> normalizers
        hipMemsetAsync(nsrc, 0, (size_t)N * 4, stream);
        hipMemsetAsync(ndst, 0, (size_t)N * 4, stream);
        hipMemsetAsync(pooled, 0, (size_t)G * DHF * 4, stream);
        hipMemsetAsync(cnt, 0, (size_t)G * 4, stream);
        k_deg<<<(E + 255) / 256, 256, 0, stream>>>(src, nsrc, E);
        k_deg<<<(E + 255) / 256, 256, 0, stream>>>(dst, ndst, E);
        k_norm<<<(N + 255) / 256, 256, 0, stream>>>(nsrc, N);
        k_norm<<<(N + 255) / 256, 256, 0, stream>>>(ndst, N);

        const float* Ws[3] = { W0, W1, W2 };
        for (int l = 0; l < 3; l++) {
            const float* Hin = (l == 0) ? x : M1;
            const int gemmGrid = (N + 127) / 128;
            if (l == 0)
                k_gemm_wmma<64><<<gemmGrid, 256, 0, stream>>>(Hin, Ws[l], nsrc, M0, N);
            else
                k_gemm_wmma<128><<<gemmGrid, 256, 0, stream>>>(Hin, Ws[l], nsrc, M0, N);

            hipMemsetAsync(M1, 0, (size_t)N * DHF * 4, stream);
            {
                long long tot = (long long)E * 32;
                k_scatter<<<(int)((tot + 255) / 256), 256, 0, stream>>>(M0, src, dst, M1, E);
            }
            {
                long long tot = (long long)N * DHF;
                k_bias_relu<<<(int)((tot + 255) / 256), 256, 0, stream>>>(M1, ndst, bb + l * DHF, N);
            }
        }
        {
            long long tot = (long long)N * 32;
            k_pool<<<(int)((tot + 255) / 256), 256, 0, stream>>>(M1, gid, pooled, cnt, N);
        }
    }

    k_mlp<<<G, 128, 0, stream>>>(pool_su, cnt_su, pool_sv, cnt_sv,
                                 (const float*)d_in[2], NGF,
                                 (const float*)d_in[17], (const float*)d_in[18],
                                 (const float*)d_in[19], (const float*)d_in[20],
                                 (const float*)d_in[21], (const float*)d_in[22],
                                 (float*)d_out);
}
